// NLGate_77498389889068
// MI455X (gfx1250) — matrix operand order repaired and checked by reading, not compiled, not run on hardware
//
#include <hip/hip_runtime.h>
#include <hip/hip_bf16.h>
#include <stdint.h>

// ---------------- problem constants ----------------
#define Bn   4
#define Cc   512
#define Dd   256
#define Tt   16
#define HWn  196
#define NQ   3136     // T*H*W
#define NKV  8192     // AT*AF
#define C2   1024     // 2*C

typedef __bf16 bf16;
typedef __attribute__((ext_vector_type(16))) __bf16        bf16x16;
typedef __attribute__((ext_vector_type(8)))  float         f32x8;
typedef __attribute__((ext_vector_type(8)))  unsigned int  u32x8;

// ---------------- helpers ----------------
// A-fragment K index for element pair vv (pairs are K,K+1 -> one dword)
__device__ __forceinline__ int akp(int vv, int hi) {
    return (vv * 2) + (hi ? 16 : 0);
}
__device__ __forceinline__ f32x8 wmma_bb(const bf16x16& a, const bf16x16& b, const f32x8& c) {
    return __builtin_amdgcn_wmma_f32_16x16x32_bf16(false, a, false, b, (short)0, c, false, false);
}
__device__ __forceinline__ f32x8 wmma_uu(const u32x8& a, const u32x8& b, const f32x8& c) {
    return wmma_bb(__builtin_bit_cast(bf16x16, a), __builtin_bit_cast(bf16x16, b), c);
}
__device__ __forceinline__ f32x8 wmma_bu(const bf16x16& a, const u32x8& b, const f32x8& c) {
    return wmma_bb(a, __builtin_bit_cast(bf16x16, b), c);
}
__device__ __forceinline__ f32x8 wmma_ub(const u32x8& a, const bf16x16& b, const f32x8& c) {
    return wmma_bb(__builtin_bit_cast(bf16x16, a), b, c);
}
__device__ __forceinline__ float redmax16(float x) {
    x = fmaxf(x, __shfl_xor(x, 1, 32));
    x = fmaxf(x, __shfl_xor(x, 2, 32));
    x = fmaxf(x, __shfl_xor(x, 4, 32));
    x = fmaxf(x, __shfl_xor(x, 8, 32));
    return x;
}
__device__ __forceinline__ float redsum16(float x) {
    x += __shfl_xor(x, 1, 32);
    x += __shfl_xor(x, 2, 32);
    x += __shfl_xor(x, 4, 32);
    x += __shfl_xor(x, 8, 32);
    return x;
}
// av[c][kv] with implicit concat: c<C -> a, c>=C -> broadcast pooled z
__device__ __forceinline__ float av_at(const float* __restrict__ a,
                                       const float* __restrict__ z,
                                       int b, int c, int kv) {
    if (c < Cc) return a[(size_t)(b * Cc + c) * NKV + kv];
    return z[(size_t)(b * Cc + (c - Cc)) * Tt + (kv >> 9)];   // t = kv/512
}

// ---------------- k0: fp32 -> bf16 weight conversion ----------------
__global__ void k_convert(const float* __restrict__ Wt, const float* __restrict__ Wp,
                          const float* __restrict__ Wg, const float* __restrict__ Wh,
                          bf16* Wtb, bf16* Wpb, bf16* Wgb, bf16* Whb) {
    int i = blockIdx.x * 256 + threadIdx.x;
    if (i < Dd * Cc)  Wtb[i] = (bf16)Wt[i];
    if (i < Dd * C2)  Wpb[i] = (bf16)Wp[i];
    if (i < Dd * C2)  Wgb[i] = (bf16)Wg[i];
    if (i < Cc * Dd)  Whb[i] = (bf16)Wh[i];
}

// ---------------- k1: z = mean_{H,W}(v) : (B,C,T) ----------------
__global__ void k_pool(const float* __restrict__ v, float* __restrict__ z) {
    int i = blockIdx.x * 256 + threadIdx.x;       // (b*C + c)*T + t
    if (i >= Bn * Cc * Tt) return;
    const float* p = v + (size_t)i * HWn;
    float s = 0.f;
#pragma unroll 4
    for (int j = 0; j < HWn; ++j) s += p[j];
    z[i] = s * (1.0f / (float)HWn);
}

// ---------------- k2: theta[b][q][d] = (Wt @ v)^T + bt  (bf16) ----------------
__global__ __launch_bounds__(256)
void k_theta(const float* __restrict__ v, const bf16* __restrict__ Wtb,
             const float* __restrict__ bt, bf16* __restrict__ thetab) {
    int wave = (int)((blockIdx.x * blockDim.x + threadIdx.x) >> 5);
    int lane = threadIdx.x & 31, hi = lane >> 4, l15 = lane & 15;
    const int tiles_pb = (NQ / 16) * (Dd / 16);   // 196*16
    int b = wave / tiles_pb;  if (b >= Bn) return;
    int t = wave % tiles_pb;
    int q0 = (t >> 4) * 16, d0 = (t & 15) * 16;
    const float* vb = v + (size_t)b * Cc * NQ;
    f32x8 acc = {0, 0, 0, 0, 0, 0, 0, 0};
    for (int c0 = 0; c0 < Cc; c0 += 32) {
        bf16x16 af;
#pragma unroll
        for (int vv = 0; vv < 8; ++vv) {
            int k = akp(vv, hi);
            af[2 * vv]     = (bf16)vb[(size_t)(c0 + k)     * NQ + q0 + l15];
            af[2 * vv + 1] = (bf16)vb[(size_t)(c0 + k + 1) * NQ + q0 + l15];
        }
        u32x8 bfr = *(const u32x8*)(Wtb + (size_t)(d0 + l15) * Cc + c0 + 16 * hi);
        acc = wmma_bu(af, bfr, acc);
    }
    float bias = bt[d0 + l15];
    bf16* op = thetab + (size_t)b * NQ * Dd;
#pragma unroll
    for (int r = 0; r < 8; ++r)
        op[(size_t)(q0 + r + 8 * hi) * Dd + d0 + l15] = (bf16)(acc[r] + bias);
}

// ---------------- k3: phiT[b][kv][d] = (Wp @ av)^T + bp  (bf16) ----------------
__global__ __launch_bounds__(256)
void k_phi(const float* __restrict__ a, const float* __restrict__ z,
           const bf16* __restrict__ Wpb, const float* __restrict__ bp_,
           bf16* __restrict__ phiT) {
    int wave = (int)((blockIdx.x * blockDim.x + threadIdx.x) >> 5);
    int lane = threadIdx.x & 31, hi = lane >> 4, l15 = lane & 15;
    const int tiles_pb = (NKV / 16) * (Dd / 16);  // 512*16
    int b = wave / tiles_pb;  if (b >= Bn) return;
    int t = wave % tiles_pb;
    int kv0 = (t >> 4) * 16, d0 = (t & 15) * 16;
    f32x8 acc = {0, 0, 0, 0, 0, 0, 0, 0};
    for (int c0 = 0; c0 < C2; c0 += 32) {
        bf16x16 af;
#pragma unroll
        for (int vv = 0; vv < 8; ++vv) {
            int k = akp(vv, hi);
            af[2 * vv]     = (bf16)av_at(a, z, b, c0 + k,     kv0 + l15);
            af[2 * vv + 1] = (bf16)av_at(a, z, b, c0 + k + 1, kv0 + l15);
        }
        u32x8 bfr = *(const u32x8*)(Wpb + (size_t)(d0 + l15) * C2 + c0 + 16 * hi);
        acc = wmma_bu(af, bfr, acc);
    }
    float bias = bp_[d0 + l15];
    bf16* op = phiT + (size_t)b * NKV * Dd;
#pragma unroll
    for (int r = 0; r < 8; ++r)
        op[(size_t)(kv0 + r + 8 * hi) * Dd + d0 + l15] = (bf16)(acc[r] + bias);
}

// ---------------- k4: gT[b][d][kv] = Wg @ av + bg  (bf16, kv-contiguous) ----------------
__global__ __launch_bounds__(256)
void k_g(const float* __restrict__ a, const float* __restrict__ z,
         const bf16* __restrict__ Wgb, const float* __restrict__ bg_,
         bf16* __restrict__ gT) {
    int wave = (int)((blockIdx.x * blockDim.x + threadIdx.x) >> 5);
    int lane = threadIdx.x & 31, hi = lane >> 4, l15 = lane & 15;
    const int tiles_pb = (Dd / 16) * (NKV / 16); // 16*512
    int b = wave / tiles_pb;  if (b >= Bn) return;
    int t = wave % tiles_pb;
    int d0 = (t >> 9) * 16, kv0 = (t & 511) * 16;
    f32x8 acc = {0, 0, 0, 0, 0, 0, 0, 0};
    for (int c0 = 0; c0 < C2; c0 += 32) {
        u32x8 af;   // A = Wg (rows d, k=c) : bf16 pairs, contiguous
#pragma unroll
        for (int vv = 0; vv < 8; ++vv)
            af[vv] = *(const unsigned int*)(Wgb + (size_t)(d0 + l15) * C2 + c0 + akp(vv, hi));
        bf16x16 bfr;   // B = av (k=c, n=kv) : strided fp32, coalesced over lanes
#pragma unroll
        for (int vv = 0; vv < 8; ++vv) {
            int c = c0 + 16 * hi + 2 * vv;
            bfr[2 * vv]     = (bf16)av_at(a, z, b, c,     kv0 + l15);
            bfr[2 * vv + 1] = (bf16)av_at(a, z, b, c + 1, kv0 + l15);
        }
        acc = wmma_ub(af, bfr, acc);
    }
    bf16* op = gT + (size_t)b * Dd * NKV;
#pragma unroll
    for (int r = 0; r < 8; ++r)
        op[(size_t)(d0 + r + 8 * hi) * NKV + kv0 + l15] = (bf16)(acc[r] + bg_[d0 + r + 8 * hi]);
}

// ---------------- k5: fused flash attention, block-level split-K ----------------
// 1 workgroup per (b, q-tile); 8 waves each own 1/8 of KV; LSE-combine in LDS.
#define KVB 64
#define PLD 66                 // padded LDS row stride (bf16 elems)
#define KVSPLIT (NKV / 8)      // 1024 columns per wave
#define SM_YP   (8 * 16 * 256 * 4)        // partial Y, f32
#define SM_PB   (8 * 16 * PLD * 2)        // P staging, bf16
#define SM_ST   (8 * 16 * 4)              // one stats array
#define SM_TOT  (SM_YP + SM_PB + 3 * SM_ST)
__global__ __launch_bounds__(256)
void k_attn(const bf16* __restrict__ thetab,
            const bf16* __restrict__ phiT,
            const bf16* __restrict__ gT,
            bf16* __restrict__ yb) {
    extern __shared__ __align__(16) char smem[];
    float* Yp  = (float*)smem;                       // [8][16][256]
    bf16*  Pball = (bf16*)(smem + SM_YP);            // [8][16*PLD]
    float* Ml  = (float*)(smem + SM_YP + SM_PB);     // [8][16]
    float* Ll  = Ml + 8 * 16;                        // [8][16]
    float* Scs = Ll + 8 * 16;                        // [8][16]

    int wid = threadIdx.x >> 5, lane = threadIdx.x & 31;
    int hi = lane >> 4, l15 = lane & 15;
    int b  = blockIdx.x / (NQ / 16);
    int qt = blockIdx.x % (NQ / 16);
    int q0 = qt * 16;
    int kvStart = wid * KVSPLIT, kvEnd = kvStart + KVSPLIT;

    // Q fragments resident: 8 x (16x32) over D=256 (same tile for all 8 waves)
    u32x8 qf[8];
    const bf16* th = thetab + ((size_t)b * NQ + q0) * Dd;
#pragma unroll
    for (int kb = 0; kb < 8; ++kb)
#pragma unroll
        for (int vv = 0; vv < 8; ++vv)
            qf[kb][vv] = *(const unsigned int*)(th + (size_t)l15 * Dd + kb * 32 + akp(vv, hi));

    f32x8 Y[16];
#pragma unroll
    for (int i = 0; i < 16; ++i) Y[i] = (f32x8){0, 0, 0, 0, 0, 0, 0, 0};
    float mrow[8], lrow[8];
#pragma unroll
    for (int r = 0; r < 8; ++r) { mrow[r] = -3.0e38f; lrow[r] = 0.f; }

    bf16* myP = Pball + wid * (16 * PLD);
    const bf16* Kb = phiT + (size_t)b * NKV * Dd;
    const bf16* Gb = gT   + (size_t)b * Dd * NKV;

    for (int kv0 = kvStart; kv0 < kvEnd; kv0 += KVB) {
        // prefetch next KV block of this split while we compute
        if (kv0 + KVB < kvEnd) {
#pragma unroll
            for (int j = 0; j < 4; ++j)
                __builtin_prefetch(Kb + (size_t)(kv0 + KVB + j * 16 + l15) * Dd + 16 * hi, 0, 3);
            __builtin_prefetch(Gb + (size_t)(lane * 8) * NKV + kv0 + KVB, 0, 3);
        }
        // ---- S = Q @ K^T  (4 tiles of 16x16, k-loop over D) ----
        f32x8 S[4];
#pragma unroll
        for (int j = 0; j < 4; ++j) {
            S[j] = (f32x8){0, 0, 0, 0, 0, 0, 0, 0};
#pragma unroll
            for (int kb = 0; kb < 8; ++kb) {
                u32x8 kf = *(const u32x8*)(Kb + (size_t)(kv0 + j * 16 + l15) * Dd + kb * 32 + 16 * hi);
                S[j] = wmma_uu(qf[kb], kf, S[j]);
            }
        }
        // ---- online softmax stats (rows live in 16-lane halves) ----
        float nm[8], sc[8];
#pragma unroll
        for (int r = 0; r < 8; ++r) {
            float bm = -3.0e38f;
#pragma unroll
            for (int j = 0; j < 4; ++j) bm = fmaxf(bm, S[j][r]);
            bm = redmax16(bm);
            nm[r] = fmaxf(mrow[r], bm);
            sc[r] = __expf(mrow[r] - nm[r]);
            mrow[r] = nm[r];
        }
        float rs[8];
#pragma unroll
        for (int r = 0; r < 8; ++r) rs[r] = 0.f;
#pragma unroll
        for (int j = 0; j < 4; ++j)
#pragma unroll
            for (int r = 0; r < 8; ++r) {
                float p = __expf(S[j][r] - nm[r]);
                rs[r] += p;
                myP[(r + 8 * hi) * PLD + j * 16 + l15] = (bf16)p;  // D-layout -> LDS
            }
#pragma unroll
        for (int r = 0; r < 8; ++r)
            lrow[r] = lrow[r] * sc[r] + redsum16(rs[r]);
        // ---- rescale Y, then Y += P @ V ----
#pragma unroll
        for (int dt = 0; dt < 16; ++dt)
#pragma unroll
            for (int r = 0; r < 8; ++r) Y[dt][r] *= sc[r];
#pragma unroll
        for (int kk = 0; kk < KVB / 32; ++kk) {
            u32x8 pf;    // A-fragment of P from LDS
#pragma unroll
            for (int vv = 0; vv < 8; ++vv)
                pf[vv] = *(const unsigned int*)(myP + l15 * PLD + kk * 32 + akp(vv, hi));
#pragma unroll
            for (int dt = 0; dt < 16; ++dt) {
                u32x8 gf = *(const u32x8*)(Gb + (size_t)(dt * 16 + l15) * NKV + kv0 + kk * 32 + 16 * hi);
                Y[dt] = wmma_uu(pf, gf, Y[dt]);
            }
        }
    }
    // ---- deposit unnormalized partials (Y_s, m_s, l_s) into LDS ----
#pragma unroll
    for (int dt = 0; dt < 16; ++dt)
#pragma unroll
        for (int r = 0; r < 8; ++r)
            Yp[((size_t)wid * 16 + r + 8 * hi) * 256 + dt * 16 + l15] = Y[dt][r];
    if (l15 == 0) {
#pragma unroll
        for (int r = 0; r < 8; ++r) {
            Ml[wid * 16 + r + 8 * hi] = mrow[r];
            Ll[wid * 16 + r + 8 * hi] = lrow[r];
        }
    }
    __syncthreads();
    // ---- LSE combine: scale_s[q] = exp(m_s - m) / l ----
    if (threadIdx.x < 16) {
        int q = threadIdx.x;
        float m = -3.0e38f;
#pragma unroll
        for (int s = 0; s < 8; ++s) m = fmaxf(m, Ml[s * 16 + q]);
        float l = 0.f;
#pragma unroll
        for (int s = 0; s < 8; ++s) l += Ll[s * 16 + q] * __expf(Ml[s * 16 + q] - m);
        float inv = 1.0f / l;
#pragma unroll
        for (int s = 0; s < 8; ++s) Scs[s * 16 + q] = __expf(Ml[s * 16 + q] - m) * inv;
    }
    __syncthreads();
    // ---- y[q][d] = sum_s Yp[s][q][d] * scale_s[q]  -> bf16, coalesced ----
    bf16* yo = yb + ((size_t)b * NQ + q0) * Dd;
    int d = threadIdx.x;   // 256 threads = one full row of D per iteration
#pragma unroll
    for (int q = 0; q < 16; ++q) {
        float acc = 0.f;
#pragma unroll
        for (int s = 0; s < 8; ++s)
            acc += Yp[((size_t)s * 16 + q) * 256 + d] * Scs[s * 16 + q];
        yo[(size_t)q * Dd + d] = (bf16)acc;
    }
}

// ---------------- k6: out = v + Wh @ y + bh ----------------
__global__ __launch_bounds__(256)
void k_h(const bf16* __restrict__ yb, const bf16* __restrict__ Whb,
         const float* __restrict__ bh_, const float* __restrict__ v,
         float* __restrict__ out) {
    int wave = (int)((blockIdx.x * blockDim.x + threadIdx.x) >> 5);
    int lane = threadIdx.x & 31, hi = lane >> 4, l15 = lane & 15;
    const int tiles_pb = (NQ / 16) * (Cc / 16); // 196*32
    int b = wave / tiles_pb;  if (b >= Bn) return;
    int t = wave % tiles_pb;
    int q0 = (t / 32) * 16, o0 = (t % 32) * 16;
    f32x8 acc = {0, 0, 0, 0, 0, 0, 0, 0};
    const bf16* ybb = yb + ((size_t)b * NQ + q0) * Dd;
    for (int d0 = 0; d0 < Dd; d0 += 32) {
        u32x8 af;
#pragma unroll
        for (int vv = 0; vv < 8; ++vv)
            af[vv] = *(const unsigned int*)(ybb + (size_t)l15 * Dd + d0 + akp(vv, hi));
        u32x8 bfr = *(const u32x8*)(Whb + (size_t)(o0 + l15) * Dd + d0 + 16 * hi);
        acc = wmma_uu(af, bfr, acc);
    }
    float bias = bh_[o0 + l15];
#pragma unroll
    for (int r = 0; r < 8; ++r) {
        size_t oi = (size_t)(b * Cc + o0 + l15) * NQ + q0 + r + 8 * hi;
        out[oi] = v[oi] + acc[r] + bias;
    }
}

// ---------------- launch ----------------
extern "C" void kernel_launch(void* const* d_in, const int* in_sizes, int n_in,
                              void* d_out, int out_size, void* d_ws, size_t ws_size,
                              hipStream_t stream) {
    const float* v  = (const float*)d_in[0];
    const float* a  = (const float*)d_in[1];
    const float* Wt = (const float*)d_in[2];
    const float* bt = (const float*)d_in[3];
    const float* Wp = (const float*)d_in[4];
    const float* bp = (const float*)d_in[5];
    const float* Wg = (const float*)d_in[6];
    const float* bg = (const float*)d_in[7];
    const float* Wh = (const float*)d_in[8];
    const float* bh = (const float*)d_in[9];
    float* out = (float*)d_out;

    char* ws = (char*)d_ws;
    size_t off = 0;
    float* z    = (float*)(ws + off); off += (size_t)Bn * Cc * Tt * 4;     // 128 KB
    bf16* Wtb   = (bf16*)(ws + off);  off += (size_t)Dd * Cc * 2;          // 256 KB
    bf16* Wpb   = (bf16*)(ws + off);  off += (size_t)Dd * C2 * 2;          // 512 KB
    bf16* Wgb   = (bf16*)(ws + off);  off += (size_t)Dd * C2 * 2;
    bf16* Whb   = (bf16*)(ws + off);  off += (size_t)Cc * Dd * 2;
    bf16* thetab= (bf16*)(ws + off);  off += (size_t)Bn * NQ * Dd * 2;     // 6.1 MB
    bf16* phiT  = (bf16*)(ws + off);  off += (size_t)Bn * NKV * Dd * 2;    // 16 MB
    bf16* gT    = (bf16*)(ws + off);  off += (size_t)Bn * Dd * NKV * 2;    // 16 MB
    bf16* yb    = (bf16*)(ws + off);  off += (size_t)Bn * NQ * Dd * 2;     // 6.1 MB

    k_convert<<<(Dd * C2 + 255) / 256, 256, 0, stream>>>(Wt, Wp, Wg, Wh, Wtb, Wpb, Wgb, Whb);
    k_pool<<<(Bn * Cc * Tt + 255) / 256, 256, 0, stream>>>(v, z);
    k_theta<<<(Bn * (NQ / 16) * (Dd / 16)) / 8, 256, 0, stream>>>(v, Wtb, bt, thetab);
    k_phi  <<<(Bn * (NKV / 16) * (Dd / 16)) / 8, 256, 0, stream>>>(a, z, Wpb, bp, phiT);
    k_g    <<<(Bn * (Dd / 16) * (NKV / 16)) / 8, 256, 0, stream>>>(a, z, Wgb, bg, gT);
    k_attn <<<Bn * (NQ / 16), 256, SM_TOT, stream>>>(thetab, phiT, gT, yb);
    k_h    <<<(Bn * (NQ / 16) * (Cc / 16)) / 8, 256, 0, stream>>>(yb, Whb, bh, v, out);
}
